// DistributedAttention_82102594830784
// MI455X (gfx1250) — matrix operand order repaired and checked by reading, not compiled, not run on hardware
//
#include <hip/hip_runtime.h>

// Problem constants (from reference): B=2, S=4096, H=1024, scale = sqrt(64) = 8.
#define BDIM 2
#define SEQ  4096
#define HID  1024
#define M2   (BDIM * SEQ)      // 8192 total rows for projections
#define SCALE 0.125f           // 1/sqrt(HEAD_DIM) = 1/8

typedef float  v8f   __attribute__((ext_vector_type(8)));
typedef __bf16 v8bf  __attribute__((ext_vector_type(8)));
typedef __bf16 v16bf __attribute__((ext_vector_type(16)));
typedef int    v4i   __attribute__((ext_vector_type(4)));

#if __has_builtin(__builtin_amdgcn_global_load_async_to_lds_b128) && \
    __has_builtin(__builtin_amdgcn_s_wait_asynccnt)
#define HAS_ASYNC_LDS 1
typedef __attribute__((address_space(1))) v4i* gptr_v4i;   // global v4i*
typedef __attribute__((address_space(3))) v4i* lptr_v4i;   // LDS v4i*
#else
#define HAS_ASYNC_LDS 0
#endif

__device__ __forceinline__ __bf16 f2bf(float f) {
  unsigned u = __builtin_bit_cast(unsigned, f);
  unsigned short r = (unsigned short)((u + 0x7FFFu + ((u >> 16) & 1u)) >> 16);
  return __builtin_bit_cast(__bf16, r);
}

__device__ __forceinline__ v16bf pack16(v8bf lo, v8bf hi) {
  v16bf a;
#pragma unroll
  for (int i = 0; i < 8; ++i) { a[i] = lo[i]; a[i + 8] = hi[i]; }
  return a;
}

// ---- cooperative LDS staging (256 threads, 128x32 bf16 tile) ----------------
// Row-major copy: Sm[r][c] = g[r*ld + c], tile 128 rows x 32 cols.
// Async path: each thread issues two GLOBAL_LOAD_ASYNC_TO_LDS_B128 (16B/lane).
__device__ __forceinline__ void stage_rowmajor(__bf16 (&Sm)[128][48],
                                               const __bf16* __restrict__ g,
                                               size_t ld, int t) {
  const int r = t >> 1, c = (t & 1) * 16;
  const __bf16* src = g + (size_t)r * ld + c;
#if HAS_ASYNC_LDS
  __builtin_amdgcn_global_load_async_to_lds_b128(
      (gptr_v4i)src, (lptr_v4i)&Sm[r][c], 0, 0);
  __builtin_amdgcn_global_load_async_to_lds_b128(
      (gptr_v4i)(src + 8), (lptr_v4i)&Sm[r][c + 8], 0, 0);
#else
  *(v8bf*)&Sm[r][c]     = *(const v8bf*)(src);
  *(v8bf*)&Sm[r][c + 8] = *(const v8bf*)(src + 8);
#endif
}

// Transposing copy: Sm[n][k] = g[k*ld + n], tile 32 k-rows x 128 n-cols.
// (async copies cannot transpose -> register path; prefetch next tile)
__device__ __forceinline__ void stage_transpose(__bf16 (&Sm)[128][48],
                                                const __bf16* __restrict__ g,
                                                size_t ld, int t) {
  const int kk = t >> 3;           // 0..31
  const int nb = (t & 7) * 16;     // 0..112
  const __bf16* src = g + (size_t)kk * ld + nb;
  v8bf x0 = *(const v8bf*)(src);
  v8bf x1 = *(const v8bf*)(src + 8);
  __builtin_prefetch(src + 32 * ld, 0, 1);   // next k-tile -> global_prefetch_b8
#pragma unroll
  for (int i = 0; i < 8; ++i) { Sm[nb + i][kk] = x0[i]; Sm[nb + 8 + i][kk] = x1[i]; }
}

// Wait for in-flight stage (async or sync) then rendezvous.
__device__ __forceinline__ void stage_fence() {
#if HAS_ASYNC_LDS
  __builtin_amdgcn_s_wait_asynccnt(0);
#endif
  __syncthreads();
}

// ---- per-wave WMMA step: 64x32 wave tile, k-step 32 -------------------------
// A fragment (16x32 bf16): lane half lh holds K chunks {lh*8..+7, lh*8+16..+23}.
// B fragment (32x16 bf16): lane = column n, K chunks {lh*8..+7, lh*8+16..+23}.
__device__ __forceinline__ void wave_mma_step(const __bf16 (&As)[128][48],
                                              const __bf16 (&Bs)[128][48],
                                              v8f (&acc)[4][2],
                                              int wm, int wn, int ln, int lh) {
  v16bf afr[4], bfr[2];
#pragma unroll
  for (int rt = 0; rt < 4; ++rt) {
    const __bf16* p = &As[wm * 64 + rt * 16 + ln][lh * 8];
    afr[rt] = pack16(*(const v8bf*)p, *(const v8bf*)(p + 16));
  }
#pragma unroll
  for (int ct = 0; ct < 2; ++ct) {
    const __bf16* p = &Bs[wn * 32 + ct * 16 + ln][lh * 8];
    bfr[ct] = pack16(*(const v8bf*)p, *(const v8bf*)(p + 16));
  }
#pragma unroll
  for (int rt = 0; rt < 4; ++rt)
#pragma unroll
    for (int ct = 0; ct < 2; ++ct)
      acc[rt][ct] = __builtin_amdgcn_wmma_f32_16x16x32_bf16(
          false, afr[rt], false, bfr[ct], (short)0, acc[rt][ct], false, false);
}

// ---- fp32 -> bf16 cast ------------------------------------------------------
__global__ void cast_f32_bf16_kernel(const float* __restrict__ s,
                                     __bf16* __restrict__ d, int n) {
  int i = (blockIdx.x * blockDim.x + threadIdx.x) * 4;
  if (i < n) {
    float4 v = *(const float4*)(s + i);
    d[i + 0] = f2bf(v.x); d[i + 1] = f2bf(v.y);
    d[i + 2] = f2bf(v.z); d[i + 3] = f2bf(v.w);
  }
}

// ---- generic bf16 GEMM: C = A * B' (+bias), double-buffered LDS -------------
// BTRANS=1: B'(k,n) = Bm[n*ldb + k]  (torch Linear weight, or K-matrix rows)
// BTRANS=0: B'(k,n) = Bm[k*ldb + n]  (plain row-major operand, e.g. V)
template <bool BTRANS, bool OUTF32, bool BIAS>
__global__ __launch_bounds__(256) void gemm_bf16_kernel(
    const __bf16* __restrict__ A, const __bf16* __restrict__ Bm,
    const float* __restrict__ bias, void* __restrict__ Cout,
    int M, int N, int K, int lda, int ldb,
    long long bsA, long long bsB, long long bsC) {
  __shared__ __bf16 As[2][128][48];
  __shared__ __bf16 Bs[2][128][48];
  const int t = threadIdx.x;
  const int lane = t & 31, wave = t >> 5;
  const int wm = wave >> 2, wn = wave & 3;
  const int ln = lane & 15, lh = lane >> 4;
  const int m0 = blockIdx.y * 128, n0 = blockIdx.x * 128;
  const int bz = blockIdx.z;
  A  += (size_t)bz * bsA;
  Bm += (size_t)bz * bsB;

  v8f acc[4][2] = {};

  // prologue: stage k-step 0 into buffer 0
  stage_rowmajor(As[0], A + (size_t)m0 * lda, lda, t);
  if (BTRANS) stage_rowmajor(Bs[0], Bm + (size_t)n0 * ldb, ldb, t);
  else        stage_transpose(Bs[0], Bm + n0, ldb, t);

  const int nk = K >> 5;
  for (int i = 0; i < nk; ++i) {
    const int cur = i & 1;
    stage_fence();
    if (i + 1 < nk) {
      const int k1 = (i + 1) << 5;
      stage_rowmajor(As[cur ^ 1], A + (size_t)m0 * lda + k1, lda, t);
      if (BTRANS) stage_rowmajor(Bs[cur ^ 1], Bm + (size_t)n0 * ldb + k1, ldb, t);
      else        stage_transpose(Bs[cur ^ 1], Bm + (size_t)k1 * ldb + n0, ldb, t);
    }
    wave_mma_step(As[cur], Bs[cur], acc, wm, wn, ln, lh);
  }

  float*  Cf = (float*)Cout;
  __bf16* Cb = (__bf16*)Cout;
#pragma unroll
  for (int rt = 0; rt < 4; ++rt)
#pragma unroll
    for (int ct = 0; ct < 2; ++ct)
#pragma unroll
      for (int r = 0; r < 8; ++r) {
        const int grow = m0 + wm * 64 + rt * 16 + r + lh * 8;
        const int gcol = n0 + wn * 32 + ct * 16 + ln;
        float v = acc[rt][ct][r];
        if (BIAS) v += bias[gcol];
        const size_t idx = (size_t)bz * bsC + (size_t)grow * N + gcol;
        if (OUTF32) Cf[idx] = v;
        else        Cb[idx] = f2bf(v);
      }
}

// ---- shared score-tile compute (128 q x 128 keys, K=HID), double-buffered ---
__device__ __forceinline__ void score_tile(const __bf16* __restrict__ Q,
                                           const __bf16* __restrict__ Km,
                                           __bf16 (&As)[2][128][48],
                                           __bf16 (&Bs)[2][128][48],
                                           v8f (&acc)[4][2],
                                           int m0, int n0, int t,
                                           int wm, int wn, int ln, int lh) {
  stage_rowmajor(As[0], Q  + (size_t)m0 * HID, HID, t);
  stage_rowmajor(Bs[0], Km + (size_t)n0 * HID, HID, t);   // B'(h,key)=K[key,h]
  const int nk = HID >> 5;
  for (int i = 0; i < nk; ++i) {
    const int cur = i & 1;
    stage_fence();
    if (i + 1 < nk) {
      const int k1 = (i + 1) << 5;
      stage_rowmajor(As[cur ^ 1], Q  + (size_t)m0 * HID + k1, HID, t);
      stage_rowmajor(Bs[cur ^ 1], Km + (size_t)n0 * HID + k1, HID, t);
    }
    wave_mma_step(As[cur], Bs[cur], acc, wm, wn, ln, lh);
  }
}

// ---- pass 1 of softmax: streaming QK^T with online row max / sum-exp --------
__global__ __launch_bounds__(256) void row_stats_kernel(
    const __bf16* __restrict__ Q, const __bf16* __restrict__ Km,
    float* __restrict__ rowM, float* __restrict__ rowL) {
  __shared__ __bf16 As[2][128][48];
  __shared__ __bf16 Bs[2][128][48];
  __shared__ float smax[4][128];
  __shared__ float ssum[4][128];
  const int t = threadIdx.x;
  const int lane = t & 31, wave = t >> 5;
  const int wm = wave >> 2, wn = wave & 3;
  const int ln = lane & 15, lh = lane >> 4;
  const int b = blockIdx.z;
  const int m0 = blockIdx.x * 128;
  Q  += (size_t)b * SEQ * HID;
  Km += (size_t)b * SEQ * HID;

  float mrun[4][8], lrun[4][8];
#pragma unroll
  for (int rt = 0; rt < 4; ++rt)
#pragma unroll
    for (int r = 0; r < 8; ++r) { mrun[rt][r] = -3.0e38f; lrun[rt][r] = 0.f; }

  for (int n0 = 0; n0 < SEQ; n0 += 128) {
    v8f acc[4][2] = {};
    score_tile(Q, Km, As, Bs, acc, m0, n0, t, wm, wn, ln, lh);
    __syncthreads();   // tile done before stats (and before next tile restages)
    // online softmax stats over this wave's 32 key columns
#pragma unroll
    for (int rt = 0; rt < 4; ++rt)
#pragma unroll
      for (int r = 0; r < 8; ++r) {
        float s0 = acc[rt][0][r] * SCALE;
        float s1 = acc[rt][1][r] * SCALE;
        float mx = fmaxf(s0, s1);
#pragma unroll
        for (int off = 1; off < 16; off <<= 1)
          mx = fmaxf(mx, __shfl_xor(mx, off, 32));   // stays in 16-lane half
        const float nm = fmaxf(mrun[rt][r], mx);
        float sum = __expf(s0 - nm) + __expf(s1 - nm);
#pragma unroll
        for (int off = 1; off < 16; off <<= 1)
          sum += __shfl_xor(sum, off, 32);
        lrun[rt][r] = lrun[rt][r] * __expf(mrun[rt][r] - nm) + sum;
        mrun[rt][r] = nm;
      }
  }

  // combine the 4 column-strip waves per row
#pragma unroll
  for (int rt = 0; rt < 4; ++rt)
#pragma unroll
    for (int r = 0; r < 8; ++r) {
      const int row = wm * 64 + rt * 16 + r + lh * 8;
      if (ln == 0) { smax[wn][row] = mrun[rt][r]; ssum[wn][row] = lrun[rt][r]; }
    }
  __syncthreads();
  if (t < 128) {
    float m = smax[0][t];
#pragma unroll
    for (int w2 = 1; w2 < 4; ++w2) m = fmaxf(m, smax[w2][t]);
    float l = 0.f;
#pragma unroll
    for (int w2 = 0; w2 < 4; ++w2) l += ssum[w2][t] * __expf(smax[w2][t] - m);
    rowM[(size_t)b * SEQ + m0 + t] = m;
    rowL[(size_t)b * SEQ + m0 + t] = l;
  }
}

// ---- pass 2: recompute scores, write normalized P in bf16 -------------------
__global__ __launch_bounds__(256) void p_write_kernel(
    const __bf16* __restrict__ Q, const __bf16* __restrict__ Km,
    const float* __restrict__ rowM, const float* __restrict__ rowL,
    __bf16* __restrict__ P) {
  __shared__ __bf16 As[2][128][48];
  __shared__ __bf16 Bs[2][128][48];
  __shared__ float sm[128];
  __shared__ float sli[128];
  const int t = threadIdx.x;
  const int lane = t & 31, wave = t >> 5;
  const int wm = wave >> 2, wn = wave & 3;
  const int ln = lane & 15, lh = lane >> 4;
  const int b = blockIdx.z;
  const int m0 = blockIdx.y * 128, n0 = blockIdx.x * 128;
  Q  += (size_t)b * SEQ * HID;
  Km += (size_t)b * SEQ * HID;
  P  += (size_t)b * SEQ * SEQ;

  if (t < 128) {
    sm[t]  = rowM[(size_t)b * SEQ + m0 + t];
    sli[t] = 1.0f / rowL[(size_t)b * SEQ + m0 + t];
  }

  v8f acc[4][2] = {};
  score_tile(Q, Km, As, Bs, acc, m0, n0, t, wm, wn, ln, lh);

#pragma unroll
  for (int rt = 0; rt < 4; ++rt)
#pragma unroll
    for (int ct = 0; ct < 2; ++ct)
#pragma unroll
      for (int r = 0; r < 8; ++r) {
        const int lrow = wm * 64 + rt * 16 + r + lh * 8;
        const int gcol = n0 + wn * 32 + ct * 16 + ln;
        const float s = acc[rt][ct][r] * SCALE;
        const float p = __expf(s - sm[lrow]) * sli[lrow];
        P[(size_t)(m0 + lrow) * SEQ + gcol] = f2bf(p);
      }
}

// ---- host-side orchestration ------------------------------------------------
extern "C" void kernel_launch(void* const* d_in, const int* in_sizes, int n_in,
                              void* d_out, int out_size, void* d_ws, size_t ws_size,
                              hipStream_t stream) {
  const float* x  = (const float*)d_in[0];
  const float* Wq = (const float*)d_in[1];
  const float* bq = (const float*)d_in[2];
  const float* Wk = (const float*)d_in[3];
  const float* bk = (const float*)d_in[4];
  const float* Wv = (const float*)d_in[5];
  const float* bv = (const float*)d_in[6];
  const float* Wo = (const float*)d_in[7];
  const float* bo = (const float*)d_in[8];
  float* out = (float*)d_out;

  char* wsp = (char*)d_ws;
  size_t off = 0;
  auto alloc = [&](size_t bytes) {
    char* p = wsp + off;
    off += (bytes + 255) & ~(size_t)255;
    return (void*)p;
  };
  __bf16* xb    = (__bf16*)alloc((size_t)M2 * HID * 2);
  __bf16* Wqb   = (__bf16*)alloc((size_t)HID * HID * 2);
  __bf16* Wkb   = (__bf16*)alloc((size_t)HID * HID * 2);
  __bf16* Wvb   = (__bf16*)alloc((size_t)HID * HID * 2);
  __bf16* Wob   = (__bf16*)alloc((size_t)HID * HID * 2);
  __bf16* Qb    = (__bf16*)alloc((size_t)M2 * HID * 2);
  __bf16* Kb    = (__bf16*)alloc((size_t)M2 * HID * 2);
  __bf16* Vb    = (__bf16*)alloc((size_t)M2 * HID * 2);
  __bf16* attnb = (__bf16*)alloc((size_t)M2 * HID * 2);
  float*  rowM  = (float*)alloc((size_t)M2 * 4);
  float*  rowL  = (float*)alloc((size_t)M2 * 4);
  __bf16* Pb    = (__bf16*)alloc((size_t)BDIM * SEQ * SEQ * 2);

  const dim3 blk(256);

  // Casts to bf16
  {
    int n = M2 * HID;
    cast_f32_bf16_kernel<<<dim3((n / 4 + 255) / 256), blk, 0, stream>>>(x, xb, n);
    int nw = HID * HID;
    dim3 gw((nw / 4 + 255) / 256);
    cast_f32_bf16_kernel<<<gw, blk, 0, stream>>>(Wq, Wqb, nw);
    cast_f32_bf16_kernel<<<gw, blk, 0, stream>>>(Wk, Wkb, nw);
    cast_f32_bf16_kernel<<<gw, blk, 0, stream>>>(Wv, Wvb, nw);
    cast_f32_bf16_kernel<<<gw, blk, 0, stream>>>(Wo, Wob, nw);
  }

  // QKV projections: [8192,1024] = xb[8192,1024] x W^T (+bias), bf16 out
  {
    dim3 g(HID / 128, M2 / 128, 1);
    gemm_bf16_kernel<true, false, true><<<g, blk, 0, stream>>>(
        xb, Wqb, bq, Qb, M2, HID, HID, HID, HID, 0, 0, 0);
    gemm_bf16_kernel<true, false, true><<<g, blk, 0, stream>>>(
        xb, Wkb, bk, Kb, M2, HID, HID, HID, HID, 0, 0, 0);
    gemm_bf16_kernel<true, false, true><<<g, blk, 0, stream>>>(
        xb, Wvb, bv, Vb, M2, HID, HID, HID, HID, 0, 0, 0);
  }

  // Softmax stats (streaming QK^T, online max/sum)
  {
    dim3 g(SEQ / 128, 1, BDIM);
    row_stats_kernel<<<g, blk, 0, stream>>>(Qb, Kb, rowM, rowL);
  }

  // Normalized probabilities P (bf16, 67 MB -> lives in L2)
  {
    dim3 g(SEQ / 128, SEQ / 128, BDIM);
    p_write_kernel<<<g, blk, 0, stream>>>(Qb, Kb, rowM, rowL, Pb);
  }

  // attn = P x V   (per batch: M=4096, N=1024, K=4096)
  {
    dim3 g(HID / 128, SEQ / 128, BDIM);
    gemm_bf16_kernel<false, false, false><<<g, blk, 0, stream>>>(
        Pb, Vb, nullptr, attnb, SEQ, HID, SEQ, SEQ, HID,
        (long long)SEQ * SEQ, (long long)SEQ * HID, (long long)SEQ * HID);
  }

  // out = attn x Wo^T + bo, fp32 output
  {
    dim3 g(HID / 128, M2 / 128, 1);
    gemm_bf16_kernel<true, true, true><<<g, blk, 0, stream>>>(
        attnb, Wob, bo, out, M2, HID, HID, HID, HID, 0, 0, 0);
  }
}
